// Tr_Decoder_54322746360070
// MI455X (gfx1250) — matrix operand order repaired and checked by reading, not compiled, not run on hardware
//
#include <hip/hip_runtime.h>
#include <hip/hip_bf16.h>

// ---------------------------------------------------------------------------
// Transformer decoder (Q=12 steps, L=4 layers, B=512, D=H=512, heads=8x64,
// cross-attn source S=48) for gfx1250 (MI455X).
// Big GEMMs -> v_wmma_f32_16x16x32_f16 (f16 inputs, f32 accum), 32x64 wave
// tiles (2x4 accumulators) for 1.5 loads/WMMA arithmetic intensity.
// Cross-attn K/V hoisted out of the step loop (computed once per layer).
// ---------------------------------------------------------------------------

typedef __attribute__((ext_vector_type(16))) _Float16 v16h;
typedef __attribute__((ext_vector_type(8)))  float    v8f;

union Frag {
    v16h  h;
    uint4 u[2];
};

static constexpr int Bsz  = 512;   // batch
static constexpr int Dm   = 512;   // model dim (= H)
static constexpr int Lnum = 4;     // layers
static constexpr int NH   = 8;     // heads
static constexpr int HD   = 64;    // head dim
static constexpr int Qs   = 12;    // decode steps
static constexpr int Slen = 48;    // cross-attn source length

// ---------------------------------------------------------------------------
// Weight convert: (L, D, H) f32 row-major -> per-(layer,slot) f16 [N][K]
// (transposed so B-matrix WMMA fragments are two contiguous 16B reads).
// ---------------------------------------------------------------------------
__global__ __launch_bounds__(256) void wcvt_kernel(const float* __restrict__ w,
                                                   _Float16* __restrict__ wt,
                                                   int slot) {
    int idx = blockIdx.x * 256 + threadIdx.x;       // over L*D*H = 4*512*512
    int l  = idx >> 18;
    int r  = idx & 262143;
    int kk = r >> 9;
    int n  = r & 511;
    wt[((size_t)(l * 10 + slot) << 18) + ((size_t)n << 9) + kk] = (_Float16)w[idx];
}

// f32 -> f16 elementwise (activation staging)
__global__ __launch_bounds__(256) void cvt16_kernel(const float* __restrict__ in,
                                                    _Float16* __restrict__ out,
                                                    int count) {
    int idx = blockIdx.x * 256 + threadIdx.x;
    if (idx < count) out[idx] = (_Float16)in[idx];
}

// zero y[:, 0, :]
__global__ __launch_bounds__(256) void zero_y0_kernel(float* __restrict__ y) {
    int idx = blockIdx.x * 256 + threadIdx.x;       // over B*512
    int dd = idx & 511;
    int b  = idx >> 9;
    y[(size_t)b * (Qs * Dm) + dd] = 0.0f;
}

// ---------------------------------------------------------------------------
// WMMA GEMM:  out = A(MxK) @ Bt(NxK)^T + bias  [+relu]
// A: f16 row-major [M][K]; Bt: f16 [N][K]; bias f32 [N].
// Block 256 thr = 8 waves (4 m x 2 n); wave -> 32x64 tile (2x4 accumulators).
// Block tile 128x128.  M%128==0, N%128==0, K%32==0 guaranteed by caller.
// ---------------------------------------------------------------------------
__global__ __launch_bounds__(256) void gemm_wmma_kernel(
    const _Float16* __restrict__ A, const _Float16* __restrict__ Bt,
    const float* __restrict__ bias,
    float* __restrict__ outF, _Float16* __restrict__ outH,
    int M, int N, int Kd, int relu) {
    const int lane = threadIdx.x & 31;
    const int wave = threadIdx.x >> 5;
    const int wm = wave & 3;             // 0..3
    const int wn = wave >> 2;            // 0..1
    const int mbase = blockIdx.x * 128 + wm * 32;
    const int nbase = blockIdx.y * 128 + wn * 64;
    const int lm   = lane & 15;
    const int lk8  = (lane >> 4) * 8;    // A fragment k offset (ISA 16-bit A layout)
    const int lk16 = (lane >> 4) * 16;   // unused (B fragments use lk8, as A does)

    const _Float16* Arow0 = A + (size_t)(mbase + lm)      * Kd;
    const _Float16* Arow1 = A + (size_t)(mbase + 16 + lm) * Kd;
    const _Float16* Brow[4];
    for (int j = 0; j < 4; ++j)
        Brow[j] = Bt + (size_t)(nbase + j * 16 + lm) * Kd;

    v8f acc[2][4] = {};

    for (int k0 = 0; k0 < Kd; k0 += 32) {
        // prefetch a few tiles ahead into the near (WGP) cache
        __builtin_prefetch(Arow0 + k0 + 128, 0, 3);
        __builtin_prefetch(Brow[0] + k0 + 128, 0, 3);

        Frag a0, a1, b[4];
        // A 16x32 frag: halves 0..7 -> k = lk8..lk8+7 ; halves 8..15 -> k = lk8+16..lk8+23
        a0.u[0] = *(const uint4*)(Arow0 + k0 + lk8);
        a0.u[1] = *(const uint4*)(Arow0 + k0 + lk8 + 16);
        a1.u[0] = *(const uint4*)(Arow1 + k0 + lk8);
        a1.u[1] = *(const uint4*)(Arow1 + k0 + lk8 + 16);
        // B 32x16 frags: same k order as A (lk8..lk8+7, lk8+16..lk8+23)
        for (int j = 0; j < 4; ++j) {
            b[j].u[0] = *(const uint4*)(Brow[j] + k0 + lk8);
            b[j].u[1] = *(const uint4*)(Brow[j] + k0 + lk8 + 16);
        }
        for (int j = 0; j < 4; ++j) {
            acc[0][j] = __builtin_amdgcn_wmma_f32_16x16x32_f16(
                false, a0.h, false, b[j].h, (short)0, acc[0][j], false, false);
            acc[1][j] = __builtin_amdgcn_wmma_f32_16x16x32_f16(
                false, a1.h, false, b[j].h, (short)0, acc[1][j], false, false);
        }
    }

    // C/D layout: VGPR r, lanes 0-15 -> M=r, lanes 16-31 -> M=r+8 ; N = lane&15
    const int rofs = (lane >> 4) * 8;
    for (int i = 0; i < 2; ++i) {
        for (int j = 0; j < 4; ++j) {
            const int col = nbase + j * 16 + lm;
            const float bv = bias[col];
            for (int r = 0; r < 8; ++r) {
                const int row = mbase + i * 16 + r + rofs;
                float v = acc[i][j][r] + bv;
                if (relu) v = fmaxf(v, 0.0f);
                const size_t o = (size_t)row * N + col;
                if (outF) outF[o] = v;
                if (outH) outH[o] = (_Float16)v;
            }
        }
    }
}

// ---------------------------------------------------------------------------
// Attention scores + softmax: one wave per (b, head, query row i), 8 waves
// per 256-thread block for WGP occupancy.
// q: f32 (B*tq, 512) head-major; k: f16 (B*tk, 512); probs: (B*NH*tq, tk) f32.
// ---------------------------------------------------------------------------
__global__ __launch_bounds__(256) void attn_softmax_kernel(
    const float* __restrict__ q, const _Float16* __restrict__ k,
    float* __restrict__ probs, int tq, int tk, float scale) {
    const int gid = blockIdx.x * 8 + (threadIdx.x >> 5);   // (b*NH + h)*tq + i
    const int lane = threadIdx.x & 31;
    const int i  = gid % tq;
    const int bh = gid / tq;
    const int h  = bh & (NH - 1);
    const int b  = bh >> 3;

    const float* qp = q + ((size_t)(b * tq + i)) * Dm + h * HD;
    const int j0 = lane, j1 = lane + 32;
    float s0 = -1e30f, s1 = -1e30f;
    if (j0 < tk) {
        const _Float16* kp = k + ((size_t)(b * tk + j0)) * Dm + h * HD;
        float acc = 0.0f;
        for (int dd = 0; dd < HD; ++dd) acc += qp[dd] * (float)kp[dd];
        s0 = acc * scale;
    }
    if (j1 < tk) {
        const _Float16* kp = k + ((size_t)(b * tk + j1)) * Dm + h * HD;
        float acc = 0.0f;
        for (int dd = 0; dd < HD; ++dd) acc += qp[dd] * (float)kp[dd];
        s1 = acc * scale;
    }
    float m = fmaxf(s0, s1);
    for (int off = 16; off > 0; off >>= 1) m = fmaxf(m, __shfl_xor(m, off, 32));
    float e0 = (j0 < tk) ? __expf(s0 - m) : 0.0f;
    float e1 = (j1 < tk) ? __expf(s1 - m) : 0.0f;
    float s = e0 + e1;
    for (int off = 16; off > 0; off >>= 1) s += __shfl_xor(s, off, 32);
    const float inv = 1.0f / s;
    float* pp = probs + (size_t)gid * tk;
    if (j0 < tk) pp[j0] = e0 * inv;
    if (j1 < tk) pp[j1] = e1 * inv;
}

// probs @ V -> f16 output, thread per (b, i, h, dd)
__global__ __launch_bounds__(256) void attn_av_kernel(
    const float* __restrict__ probs, const _Float16* __restrict__ v,
    _Float16* __restrict__ o, int tq, int tk) {
    const int idx = blockIdx.x * 256 + threadIdx.x;  // over B*tq*512
    const int dd  = idx & 63;
    const int h   = (idx >> 6) & 7;
    const int row = idx >> 9;                         // b*tq + i
    const int b   = row / tq;
    const int i   = row - b * tq;
    const float*    pp = probs + ((size_t)((b * NH + h) * tq + i)) * tk;
    const _Float16* vp = v + ((size_t)b * tk) * Dm + h * HD + dd;
    float acc = 0.0f;
    for (int j = 0; j < tk; ++j) acc += pp[j] * (float)vp[(size_t)j * Dm];
    o[idx] = (_Float16)acc;
}

// ---------------------------------------------------------------------------
// LayerNorm over D=512 with fused residual; writes f32 and f16 copies.
// One block (256 thr) per row; each thread owns 2 elements.
// ---------------------------------------------------------------------------
__global__ __launch_bounds__(256) void ln_kernel(
    const float* __restrict__ a, const float* __restrict__ res,
    const float* __restrict__ g, const float* __restrict__ bb,
    float* __restrict__ out, _Float16* __restrict__ oh) {
    __shared__ float red[256];
    const int row = blockIdx.x;
    const int tid = threadIdx.x;
    const size_t base = (size_t)row * Dm;
    float v0 = a[base + tid]       + res[base + tid];
    float v1 = a[base + tid + 256] + res[base + tid + 256];
    red[tid] = v0 + v1;
    __syncthreads();
    for (int off = 128; off > 0; off >>= 1) {
        if (tid < off) red[tid] += red[tid + off];
        __syncthreads();
    }
    const float mean = red[0] * (1.0f / Dm);
    __syncthreads();
    const float d0 = v0 - mean, d1 = v1 - mean;
    red[tid] = d0 * d0 + d1 * d1;
    __syncthreads();
    for (int off = 128; off > 0; off >>= 1) {
        if (tid < off) red[tid] += red[tid + off];
        __syncthreads();
    }
    const float rstd = rsqrtf(red[0] * (1.0f / Dm) + 1e-6f);
    const float o0 = d0 * rstd * g[tid]       + bb[tid];
    const float o1 = d1 * rstd * g[tid + 256] + bb[tid + 256];
    out[base + tid]       = o0;
    out[base + tid + 256] = o1;
    oh[base + tid]        = (_Float16)o0;
    oh[base + tid + 256]  = (_Float16)o1;
}

// h = y + positional encoding; writes f32 and f16 copies.  y: (B,12,D), h: (B,t,D)
__global__ __launch_bounds__(256) void pe_add_kernel(
    const float* __restrict__ y, float* __restrict__ h,
    _Float16* __restrict__ hh, int t) {
    const int idx = blockIdx.x * 256 + threadIdx.x;   // over B*t*512
    const int dd = idx & 511;
    const int r  = idx >> 9;                          // b*t + n
    const int b  = r / t;
    const int n  = r - b * t;
    const float ex  = 2.0f * floorf((float)dd * 0.5f) * (1.0f / Dm);
    const float ang = (float)n / __powf(10000.0f, ex);
    const float pv  = (dd & 1) ? __cosf(ang) : __sinf(ang);
    const float v = y[((size_t)b * Qs + n) * Dm + dd] + pv;
    h[idx]  = v;
    hh[idx] = (_Float16)v;
}

// token = sum_n h[b,n,:]*dense_w[n] + dense_b -> d_out[:,q,:] and y[:,q+1,:]
__global__ __launch_bounds__(256) void token_kernel(
    const float* __restrict__ h, const float* __restrict__ dw,
    const float* __restrict__ db, float* __restrict__ out,
    float* __restrict__ y, int t /* = q+1 */) {
    const int idx = blockIdx.x * 256 + threadIdx.x;   // over B*512
    const int dd = idx & 511;
    const int b  = idx >> 9;
    float acc = db[0];
    for (int n = 0; n < t; ++n) acc += h[((size_t)b * t + n) * Dm + dd] * dw[n];
    out[((size_t)b * Qs + (t - 1)) * Dm + dd] = acc;
    if (t < Qs) y[((size_t)b * Qs + t) * Dm + dd] = acc;
}

// ---------------------------------------------------------------------------
// Host orchestration
// ---------------------------------------------------------------------------
static inline char* ws_bump(char*& p, size_t bytes) {
    char* r = p;
    p += (bytes + 255) & ~(size_t)255;
    return r;
}

extern "C" void kernel_launch(void* const* d_in, const int* in_sizes, int n_in,
                              void* d_out, int out_size, void* d_ws, size_t ws_size,
                              hipStream_t stream) {
    (void)in_sizes; (void)n_in; (void)out_size; (void)ws_size;

    const float* x = (const float*)d_in[0];
    // slot order in wt buffer: 0=SAQ 1=SAK 2=SAV 3=SAO 4=CAQ 5=CAK 6=CAV 7=CAO 8=FF1 9=FF2
    const float* wsrc[10] = {
        (const float*)d_in[1],  (const float*)d_in[3],  (const float*)d_in[5],  (const float*)d_in[7],
        (const float*)d_in[9],  (const float*)d_in[11], (const float*)d_in[13], (const float*)d_in[15],
        (const float*)d_in[17], (const float*)d_in[19]
    };
    const float* bsrc[10] = {
        (const float*)d_in[2],  (const float*)d_in[4],  (const float*)d_in[6],  (const float*)d_in[8],
        (const float*)d_in[10], (const float*)d_in[12], (const float*)d_in[14], (const float*)d_in[16],
        (const float*)d_in[18], (const float*)d_in[20]
    };
    const float* lng[3] = { (const float*)d_in[21], (const float*)d_in[23], (const float*)d_in[25] };
    const float* lnb[3] = { (const float*)d_in[22], (const float*)d_in[24], (const float*)d_in[26] };
    const float* dense_w = (const float*)d_in[27];
    const float* dense_b = (const float*)d_in[28];
    float* out = (float*)d_out;

    // ---- workspace layout --------------------------------------------------
    char* p = (char*)d_ws;
    const size_t nBTD  = (size_t)Bsz * Qs * Dm;        // 3,145,728
    const size_t nBSD  = (size_t)Bsz * Slen * Dm;      // 12,582,912
    _Float16* wt    = (_Float16*)ws_bump(p, (size_t)Lnum * 10 * Dm * Dm * 2);
    float*    hbuf  = (float*)   ws_bump(p, nBTD * 4);
    _Float16* hf16  = (_Float16*)ws_bump(p, nBTD * 2);
    float*    ybuf  = (float*)   ws_bump(p, nBTD * 4);
    _Float16* xf16  = (_Float16*)ws_bump(p, nBSD * 2);
    _Float16* kx[Lnum];
    _Float16* vx[Lnum];
    for (int l = 0; l < Lnum; ++l) {
        kx[l] = (_Float16*)ws_bump(p, nBSD * 2);
        vx[l] = (_Float16*)ws_bump(p, nBSD * 2);
    }
    float*    qbuf  = (float*)   ws_bump(p, nBTD * 4);
    _Float16* kf16  = (_Float16*)ws_bump(p, nBTD * 2);
    _Float16* vf16  = (_Float16*)ws_bump(p, nBTD * 2);
    _Float16* of16  = (_Float16*)ws_bump(p, nBTD * 2);
    _Float16* ffbuf = (_Float16*)ws_bump(p, nBTD * 2);
    float*    aout  = (float*)   ws_bump(p, nBTD * 4);
    float*    probs = (float*)   ws_bump(p, (size_t)Bsz * NH * Qs * Slen * 4);

    auto wptr = [&](int l, int slot) -> _Float16* {
        return wt + ((size_t)(l * 10 + slot) << 18);
    };
    auto gemm = [&](const _Float16* A, const _Float16* Bt, const float* bias,
                    float* oF, _Float16* oH, int M, int relu) {
        dim3 g(M / 128, Dm / 128);
        gemm_wmma_kernel<<<g, 256, 0, stream>>>(A, Bt, bias, oF, oH, M, Dm, Dm, relu);
    };

    const float scale = 0.125f;  // 1/sqrt(64)

    // ---- setup: weights, x, cross K/V (once per layer) ---------------------
    zero_y0_kernel<<<(Bsz * Dm) / 256, 256, 0, stream>>>(ybuf);
    for (int s = 0; s < 10; ++s)
        wcvt_kernel<<<(Lnum * Dm * Dm) / 256, 256, 0, stream>>>(wsrc[s], wt, s);
    cvt16_kernel<<<(int)(nBSD / 256), 256, 0, stream>>>(x, xf16, (int)nBSD);
    for (int l = 0; l < Lnum; ++l) {
        gemm(xf16, wptr(l, 5), bsrc[5] + l * Dm, nullptr, kx[l], Bsz * Slen, 0);
        gemm(xf16, wptr(l, 6), bsrc[6] + l * Dm, nullptr, vx[l], Bsz * Slen, 0);
    }

    // ---- autoregressive decode ---------------------------------------------
    for (int q = 0; q < Qs; ++q) {
        const int t = q + 1;
        const int M = Bsz * t;
        pe_add_kernel<<<(M * Dm) / 256, 256, 0, stream>>>(ybuf, hbuf, hf16, t);

        for (int l = 0; l < Lnum; ++l) {
            // --- self attention ---
            gemm(hf16, wptr(l, 0), bsrc[0] + l * Dm, qbuf, nullptr, M, 0);
            gemm(hf16, wptr(l, 1), bsrc[1] + l * Dm, nullptr, kf16, M, 0);
            gemm(hf16, wptr(l, 2), bsrc[2] + l * Dm, nullptr, vf16, M, 0);
            attn_softmax_kernel<<<(Bsz * NH * t) / 8, 256, 0, stream>>>(qbuf, kf16, probs, t, t, scale);
            attn_av_kernel<<<(M * Dm) / 256, 256, 0, stream>>>(probs, vf16, of16, t, t);
            gemm(of16, wptr(l, 3), bsrc[3] + l * Dm, aout, nullptr, M, 0);
            ln_kernel<<<M, 256, 0, stream>>>(aout, hbuf, lng[0] + l * Dm, lnb[0] + l * Dm, hbuf, hf16);

            // --- cross attention (K/V precomputed per layer) ---
            gemm(hf16, wptr(l, 4), bsrc[4] + l * Dm, qbuf, nullptr, M, 0);
            attn_softmax_kernel<<<(Bsz * NH * t) / 8, 256, 0, stream>>>(qbuf, kx[l], probs, t, Slen, scale);
            attn_av_kernel<<<(M * Dm) / 256, 256, 0, stream>>>(probs, vx[l], of16, t, Slen);
            gemm(of16, wptr(l, 7), bsrc[7] + l * Dm, aout, nullptr, M, 0);
            ln_kernel<<<M, 256, 0, stream>>>(aout, hbuf, lng[1] + l * Dm, lnb[1] + l * Dm, hbuf, hf16);

            // --- FFN ---
            gemm(hf16, wptr(l, 8), bsrc[8] + l * Dm, nullptr, ffbuf, M, 1);
            gemm(ffbuf, wptr(l, 9), bsrc[9] + l * Dm, aout, nullptr, M, 0);
            ln_kernel<<<M, 256, 0, stream>>>(aout, hbuf, lng[2] + l * Dm, lnb[2] + l * Dm, hbuf, hf16);
        }

        token_kernel<<<(Bsz * Dm) / 256, 256, 0, stream>>>(hbuf, dense_w, dense_b, out, ybuf, t);
    }
}
